// MPNNetDrop_26207890440264
// MI455X (gfx1250) — matrix operand order repaired and checked by reading, not compiled, not run on hardware
//
#include <hip/hip_runtime.h>
#include <hip/hip_bf16.h>
#include <math.h>

// ---------------- problem constants ----------------
#define N_CNT 16384
#define E_CNT 32768
#define B_CNT 512
#define DIM_C 64
#define HID_C 128
#define OUTC  4096   // DIM*DIM

typedef __attribute__((ext_vector_type(16))) _Float16 v16h;
typedef __attribute__((ext_vector_type(8)))  float    v8f;
typedef __attribute__((ext_vector_type(2)))  _Float16 h2;

// ---------------- WMMA fragment helpers (CDNA5 gfx1250, wave32) ----------------
// A fragment (16x32 f16, M x K): lane l holds row m=l&15, half=l>>4.
// element i: k = half*16 + i
__device__ __forceinline__ v16h load_a_frag(const _Float16* base, int ld, int lane) {
  int m = lane & 15, half = lane >> 4;
  const _Float16* rowp = base + m * ld + half * 16;
  v16h a;
#pragma unroll
  for (int i = 0; i < 8; ++i) a[i] = rowp[i];
#pragma unroll
  for (int i = 0; i < 8; ++i) a[i + 8] = rowp[8 + i];
  return a;
}

// B fragment (32x16 f16, K x N): lane l holds col n=l&15, half=l>>4.
// element i: k = half*16 + i  -> contiguous when weights stored col-major [n][k]
__device__ __forceinline__ v16h load_b_frag(const _Float16* wcol, int col_base, int K,
                                            int kbase, int lane) {
  int n = lane & 15, half = lane >> 4;
  return *(const v16h*)(wcol + (size_t)(col_base + n) * K + kbase + half * 16);
}

__device__ __forceinline__ v8f wmma16(v16h a, v16h b, v8f c) {
  return __builtin_amdgcn_wmma_f32_16x16x32_f16(false, a, false, b, (short)0, c, false, false);
}

__device__ __forceinline__ float sigm(float x) { return 1.0f / (1.0f + __expf(-x)); }

__device__ __forceinline__ void atomicMaxF(float* addr, float val) {
  if (val >= 0.0f) atomicMax((int*)addr, __float_as_int(val));
  else             atomicMin((unsigned int*)addr, __float_as_uint(val));
}

// ---------------- small utility kernels ----------------
__global__ void fill_f32(float* p, float v, int n) {
  int i = blockIdx.x * blockDim.x + threadIdx.x;
  if (i < n) p[i] = v;
}

__global__ void convert_f16(const float* in, _Float16* out, int n) {
  int i = blockIdx.x * blockDim.x + threadIdx.x;
  if (i < n) out[i] = (_Float16)in[i];
}

// out[c*rows + r] = in[r*cols + c]  (store K-major per output column)
__global__ void transpose_f16(const float* in, _Float16* out, int rows, int cols) {
  int i = blockIdx.x * blockDim.x + threadIdx.x;
  if (i >= rows * cols) return;
  int r = i / cols, c = i % cols;
  out[(size_t)c * rows + r] = (_Float16)in[i];
}

// ---------------- lin0: h = relu(x @ W[14,64] + b) ----------------
__global__ void lin0_kernel(const float* __restrict__ x, const float* __restrict__ w,
                            const float* __restrict__ b, float* __restrict__ h) {
  int i = blockIdx.x * blockDim.x + threadIdx.x;
  if (i >= N_CNT * DIM_C) return;
  int n = i >> 6, c = i & 63;
  float a = b[c];
#pragma unroll
  for (int k = 0; k < 14; ++k) a += x[n * 14 + k] * w[k * DIM_C + c];
  h[i] = fmaxf(a, 0.0f);
}

// ---------------- edge MLP stage 1: hid = relu(ea @ nn1_w + b1), f16 ----------------
__global__ void edge_mlp1_kernel(const float* __restrict__ ea, const float* __restrict__ w1,
                                 const float* __restrict__ b1, _Float16* __restrict__ hid) {
  int i = blockIdx.x * blockDim.x + threadIdx.x;
  if (i >= E_CNT * HID_C) return;
  int e = i >> 7, hcol = i & 127;
  float a = b1[hcol];
#pragma unroll
  for (int k = 0; k < 4; ++k) a += ea[e * 4 + k] * w1[k * HID_C + hcol];
  hid[i] = (_Float16)fmaxf(a, 0.0f);
}

// ---------------- edge MLP stage 2 (WMMA GEMM): ew[E,4096] = hid[E,128] @ nn2_w + b2 ----------------
// block = 256 threads (8 waves). 16 edges x 128 cols per block.
__global__ void __launch_bounds__(256) edge_mlp2_kernel(const _Float16* __restrict__ hid,
                                                        const _Float16* __restrict__ w2c,
                                                        const float* __restrict__ b2,
                                                        _Float16* __restrict__ ew) {
  __shared__ __align__(16) _Float16 Alds[16 * HID_C];
  int etile = blockIdx.x >> 5;   // E/16 = 2048 tiles
  int cgrp  = blockIdx.x & 31;   // 32 groups of 128 cols
  int eb = etile * 16;
  // A tile is 16*128 contiguous halves = 4 KB: one uint4 per thread
  ((uint4*)Alds)[threadIdx.x] = ((const uint4*)(hid + (size_t)eb * HID_C))[threadIdx.x];
  __syncthreads();
  int wave = threadIdx.x >> 5, lane = threadIdx.x & 31;
  int colbase = cgrp * 128 + wave * 16;
  v8f acc = {};
#pragma unroll
  for (int ks = 0; ks < 4; ++ks) {
    v16h a = load_a_frag(Alds + ks * 32, HID_C, lane);
    v16h b = load_b_frag(w2c, colbase, HID_C, ks * 32, lane);
    acc = wmma16(a, b, acc);
  }
  int n = lane & 15, half = lane >> 4;
  int c = colbase + n;
  float bias = b2[c];
#pragma unroll
  for (int r = 0; r < 8; ++r) {
    int m = r + 8 * half;
    ew[(size_t)(eb + m) * OUTC + c] = (_Float16)(acc[r] + bias);
  }
}

// ---------------- degree via atomics ----------------
__global__ void deg_kernel(const int* __restrict__ dst, float* __restrict__ deg) {
  int e = blockIdx.x * blockDim.x + threadIdx.x;
  if (e < E_CNT) atomicAdd(&deg[dst[e]], 1.0f);
}

// ---------------- per-edge matvec + scatter: agg[dst] += h[src] @ ew[e] ----------------
// one wave per edge; lane owns output cols {2*lane, 2*lane+1}
__global__ void __launch_bounds__(256) msg_scatter_kernel(const float* __restrict__ h,
                                                          const _Float16* __restrict__ ew,
                                                          const int* __restrict__ src,
                                                          const int* __restrict__ dst,
                                                          float* __restrict__ agg) {
  int gid = blockIdx.x * blockDim.x + threadIdx.x;
  int e = gid >> 5, lane = gid & 31;
  if (e >= E_CNT) return;
  int s = src[e], d = dst[e];
  const float* hs = h + (size_t)s * DIM_C;
  float va = hs[lane], vb = hs[lane + 32];
  const _Float16* M = ew + (size_t)e * OUTC;
  __builtin_prefetch(M + OUTC, 0, 0);  // stream next edge matrix (global_prefetch_b8)
  float acc0 = 0.0f, acc1 = 0.0f;
#pragma unroll 8
  for (int dd = 0; dd < 32; ++dd) {
    float x0 = __shfl(va, dd, 32);
    h2 m = *(const h2*)(M + dd * DIM_C + 2 * lane);
    acc0 += x0 * (float)m[0];
    acc1 += x0 * (float)m[1];
  }
#pragma unroll 8
  for (int dd = 0; dd < 32; ++dd) {
    float x1 = __shfl(vb, dd, 32);
    h2 m = *(const h2*)(M + (dd + 32) * DIM_C + 2 * lane);
    acc0 += x1 * (float)m[0];
    acc1 += x1 * (float)m[1];
  }
  atomicAdd(&agg[(size_t)d * DIM_C + 2 * lane], acc0);
  atomicAdd(&agg[(size_t)d * DIM_C + 2 * lane + 1], acc1);
}

// ---------------- fused root-matmul + GRU cell (WMMA), in-place h update ----------------
// block = 128 threads (4 waves), one 16-node tile per block
__global__ void __launch_bounds__(128) root_gru_kernel(float* __restrict__ h,
                                                       const float* __restrict__ agg,
                                                       const float* __restrict__ deg,
                                                       const _Float16* __restrict__ rwc,
                                                       const float* __restrict__ convb,
                                                       const _Float16* __restrict__ wihc,
                                                       const _Float16* __restrict__ whhc,
                                                       const float* __restrict__ bih,
                                                       const float* __restrict__ bhh) {
  __shared__ __align__(16) _Float16 hf[16 * DIM_C];
  __shared__ __align__(16) _Float16 mf[16 * DIM_C];
  __shared__ float h32[16 * DIM_C];
  __shared__ float gi[16 * 192];
  __shared__ float gh[16 * 192];
  int nb = blockIdx.x * 16;
  int t = threadIdx.x;
#pragma unroll
  for (int q = 0; q < 8; ++q) {
    int idx = t + q * 128;
    float v = h[(size_t)nb * DIM_C + idx];
    h32[idx] = v;
    hf[idx] = (_Float16)v;
  }
  __syncthreads();
  int wave = t >> 5, lane = t & 31;
  int n = lane & 15, half = lane >> 4;
  // m = relu(h @ root_w + agg/deg + conv_b) ; wave w -> cols [16w,16w+16)
  {
    int cb = wave * 16;
    v8f acc = {};
#pragma unroll
    for (int ks = 0; ks < 2; ++ks) {
      v16h a = load_a_frag(hf + ks * 32, DIM_C, lane);
      v16h b = load_b_frag(rwc, cb, DIM_C, ks * 32, lane);
      acc = wmma16(a, b, acc);
    }
    int c = cb + n;
#pragma unroll
    for (int r = 0; r < 8; ++r) {
      int m = r + 8 * half;
      int node = nb + m;
      float dv = fmaxf(deg[node], 1.0f);
      float val = acc[r] + agg[(size_t)node * DIM_C + c] / dv + convb[c];
      mf[m * DIM_C + c] = (_Float16)fmaxf(val, 0.0f);
    }
  }
  __syncthreads();
  // gi = m @ Wih (64x192), gh = h @ Whh (64x192); wave handles col tiles w, w+4, w+8
#pragma unroll
  for (int j = 0; j < 3; ++j) {
    int cb = (wave + 4 * j) * 16;
    v8f ai = {}, ah = {};
#pragma unroll
    for (int ks = 0; ks < 2; ++ks) {
      v16h am = load_a_frag(mf + ks * 32, DIM_C, lane);
      v16h ax = load_a_frag(hf + ks * 32, DIM_C, lane);
      v16h bi = load_b_frag(wihc, cb, DIM_C, ks * 32, lane);
      v16h bh = load_b_frag(whhc, cb, DIM_C, ks * 32, lane);
      ai = wmma16(am, bi, ai);
      ah = wmma16(ax, bh, ah);
    }
    int c = cb + n;
#pragma unroll
    for (int r = 0; r < 8; ++r) {
      int m = r + 8 * half;
      gi[m * 192 + c] = ai[r] + bih[c];
      gh[m * 192 + c] = ah[r] + bhh[c];
    }
  }
  __syncthreads();
  // GRU elementwise
#pragma unroll
  for (int q = 0; q < 8; ++q) {
    int idx = t + q * 128;
    int row = idx >> 6, u = idx & 63;
    float rg = sigm(gi[row * 192 + u] + gh[row * 192 + u]);
    float zg = sigm(gi[row * 192 + 64 + u] + gh[row * 192 + 64 + u]);
    float ng = tanhf(gi[row * 192 + 128 + u] + rg * gh[row * 192 + 128 + u]);
    h[(size_t)(nb + row) * DIM_C + u] = (1.0f - zg) * ng + zg * h32[row * DIM_C + u];
  }
}

// ---------------- Set2Set: LSTM cell (B=512, tiny -> scalar) ----------------
__global__ void lstm_kernel(const float* __restrict__ qstar, const float* __restrict__ hl_in,
                            float* __restrict__ hl_out, float* __restrict__ cl,
                            const float* __restrict__ wih, const float* __restrict__ whh,
                            const float* __restrict__ bih, const float* __restrict__ bhh) {
  int i = blockIdx.x * blockDim.x + threadIdx.x;
  if (i >= B_CNT * DIM_C) return;
  int b = i >> 6, u = i & 63;
  float g[4];
#pragma unroll
  for (int gidx = 0; gidx < 4; ++gidx) {
    int row = gidx * DIM_C + u;
    float a = bih[row] + bhh[row];
    for (int k = 0; k < 2 * DIM_C; ++k) a += qstar[b * 2 * DIM_C + k] * wih[row * 2 * DIM_C + k];
    for (int k = 0; k < DIM_C; ++k)     a += hl_in[b * DIM_C + k] * whh[row * DIM_C + k];
    g[gidx] = a;
  }
  float c_ = sigm(g[1]) * cl[i] + sigm(g[0]) * tanhf(g[2]);
  cl[i] = c_;
  hl_out[i] = sigm(g[3]) * tanhf(c_);
}

// ---------------- attention: e[n] = <h[n], q[batch[n]]>, segment max ----------------
__global__ void attn_e_kernel(const float* __restrict__ h, const float* __restrict__ q,
                              const int* __restrict__ batch, float* __restrict__ e,
                              float* __restrict__ emax) {
  int nidx = blockIdx.x * blockDim.x + threadIdx.x;
  if (nidx >= N_CNT) return;
  int b = batch[nidx];
  float a = 0.0f;
#pragma unroll 8
  for (int f = 0; f < DIM_C; ++f) a += h[(size_t)nidx * DIM_C + f] * q[b * DIM_C + f];
  e[nidx] = a;
  atomicMaxF(&emax[b], a);
}

__global__ void attn_exp_kernel(const float* __restrict__ e, const float* __restrict__ emax,
                                const int* __restrict__ batch, float* __restrict__ ex,
                                float* __restrict__ esum) {
  int nidx = blockIdx.x * blockDim.x + threadIdx.x;
  if (nidx >= N_CNT) return;
  int b = batch[nidx];
  float v = __expf(e[nidx] - emax[b]);
  ex[nidx] = v;
  atomicAdd(&esum[b], v);
}

__global__ void attn_r_kernel(const float* __restrict__ h, const float* __restrict__ ex,
                              const float* __restrict__ esum, const int* __restrict__ batch,
                              float* __restrict__ r) {
  int i = blockIdx.x * blockDim.x + threadIdx.x;
  if (i >= N_CNT * DIM_C) return;
  int nidx = i >> 6, f = i & 63;
  int b = batch[nidx];
  float a = ex[nidx] / fmaxf(esum[b], 1e-16f);
  atomicAdd(&r[(size_t)b * DIM_C + f], a * h[i]);
}

__global__ void qstar_kernel(const float* __restrict__ q, const float* __restrict__ r,
                             float* __restrict__ qstar) {
  int i = blockIdx.x * blockDim.x + threadIdx.x;
  if (i >= B_CNT * 2 * DIM_C) return;
  int b = i >> 7, k = i & 127;
  qstar[i] = (k < DIM_C) ? q[b * DIM_C + k] : r[b * DIM_C + (k - DIM_C)];
}

// ---------------- final MLP: y = relu(qstar@W1 + b1)@W2 + b2 ----------------
__global__ void final_kernel(const float* __restrict__ qstar, const float* __restrict__ w1,
                             const float* __restrict__ b1, const float* __restrict__ w2,
                             const float* __restrict__ b2, float* __restrict__ out) {
  int b = blockIdx.x * blockDim.x + threadIdx.x;
  if (b >= B_CNT) return;
  float acc = b2[0];
  for (int j = 0; j < DIM_C; ++j) {
    float a = b1[j];
    for (int k = 0; k < 2 * DIM_C; ++k) a += qstar[b * 2 * DIM_C + k] * w1[k * DIM_C + j];
    acc += fmaxf(a, 0.0f) * w2[j];
  }
  out[b] = acc;
}

// ---------------- host orchestration ----------------
static inline unsigned gsz(long long n, int b) { return (unsigned)((n + b - 1) / b); }

extern "C" void kernel_launch(void* const* d_in, const int* in_sizes, int n_in,
                              void* d_out, int out_size, void* d_ws, size_t ws_size,
                              hipStream_t stream) {
  (void)in_sizes; (void)n_in; (void)out_size; (void)ws_size;
  const float* x      = (const float*)d_in[0];
  const int*   eidx   = (const int*)d_in[1];
  const float* eattr  = (const float*)d_in[2];
  const int*   batch  = (const int*)d_in[3];
  const float* lin0_w = (const float*)d_in[6];
  const float* lin0_b = (const float*)d_in[7];
  const float* nn1_w  = (const float*)d_in[8];
  const float* nn1_b  = (const float*)d_in[9];
  const float* nn2_w  = (const float*)d_in[10];
  const float* nn2_b  = (const float*)d_in[11];
  const float* root_w = (const float*)d_in[12];
  const float* conv_b = (const float*)d_in[13];
  const float* gwih   = (const float*)d_in[14];
  const float* gwhh   = (const float*)d_in[15];
  const float* gbih   = (const float*)d_in[16];
  const float* gbhh   = (const float*)d_in[17];
  const float* lwih   = (const float*)d_in[18];
  const float* lwhh   = (const float*)d_in[19];
  const float* lbih   = (const float*)d_in[20];
  const float* lbhh   = (const float*)d_in[21];
  const float* lin1_w = (const float*)d_in[22];
  const float* lin1_b = (const float*)d_in[23];
  const float* lin2_w = (const float*)d_in[24];
  const float* lin2_b = (const float*)d_in[25];
  const int* src = eidx;
  const int* dst = eidx + E_CNT;
  float* out = (float*)d_out;

  char* ws = (char*)d_ws;
  size_t off = 0;
  auto carve = [&](size_t bytes) -> void* {
    void* p = ws + off;
    off = (off + bytes + 255) & ~(size_t)255;
    return p;
  };
  _Float16* hid   = (_Float16*)carve((size_t)E_CNT * HID_C * 2);
  _Float16* w2c   = (_Float16*)carve((size_t)OUTC * HID_C * 2);
  _Float16* ew    = (_Float16*)carve((size_t)E_CNT * OUTC * 2);
  _Float16* rwc   = (_Float16*)carve(DIM_C * DIM_C * 2);
  _Float16* wihc  = (_Float16*)carve(192 * DIM_C * 2);
  _Float16* whhc  = (_Float16*)carve(192 * DIM_C * 2);
  float* h     = (float*)carve((size_t)N_CNT * DIM_C * 4);
  float* agg   = (float*)carve((size_t)N_CNT * DIM_C * 4);
  float* deg   = (float*)carve((size_t)N_CNT * 4);
  float* hl0   = (float*)carve(B_CNT * DIM_C * 4);
  float* hl1   = (float*)carve(B_CNT * DIM_C * 4);
  float* cl    = (float*)carve(B_CNT * DIM_C * 4);
  float* qstar = (float*)carve(B_CNT * 2 * DIM_C * 4);
  float* ebuf  = (float*)carve((size_t)N_CNT * 4);
  float* exbuf = (float*)carve((size_t)N_CNT * 4);
  float* emax  = (float*)carve(B_CNT * 4);
  float* esum  = (float*)carve(B_CNT * 4);
  float* rbuf  = (float*)carve(B_CNT * DIM_C * 4);

  // ---- weight conversions (f16, B-matrix col-major = K-contiguous) ----
  transpose_f16<<<gsz((long long)HID_C * OUTC, 256), 256, 0, stream>>>(nn2_w, w2c, HID_C, OUTC);
  transpose_f16<<<gsz(DIM_C * DIM_C, 256), 256, 0, stream>>>(root_w, rwc, DIM_C, DIM_C);
  convert_f16<<<gsz(192 * DIM_C, 256), 256, 0, stream>>>(gwih, wihc, 192 * DIM_C);   // [j][d] already K-contig
  convert_f16<<<gsz(192 * DIM_C, 256), 256, 0, stream>>>(gwhh, whhc, 192 * DIM_C);

  // ---- lin0 + edge MLP ----
  lin0_kernel<<<gsz((long long)N_CNT * DIM_C, 256), 256, 0, stream>>>(x, lin0_w, lin0_b, h);
  edge_mlp1_kernel<<<gsz((long long)E_CNT * HID_C, 256), 256, 0, stream>>>(eattr, nn1_w, nn1_b, hid);
  edge_mlp2_kernel<<<(E_CNT / 16) * (OUTC / 128), 256, 0, stream>>>(hid, w2c, nn2_b, ew);

  // ---- degree ----
  fill_f32<<<gsz(N_CNT, 256), 256, 0, stream>>>(deg, 0.0f, N_CNT);
  deg_kernel<<<gsz(E_CNT, 256), 256, 0, stream>>>(dst, deg);

  // ---- 3 message-passing rounds ----
  for (int round = 0; round < 3; ++round) {
    fill_f32<<<gsz((long long)N_CNT * DIM_C, 256), 256, 0, stream>>>(agg, 0.0f, N_CNT * DIM_C);
    msg_scatter_kernel<<<gsz((long long)E_CNT * 32, 256), 256, 0, stream>>>(h, ew, src, dst, agg);
    root_gru_kernel<<<N_CNT / 16, 128, 0, stream>>>(h, agg, deg, rwc, conv_b, wihc, whhc, gbih, gbhh);
  }

  // ---- Set2Set (3 steps) ----
  fill_f32<<<gsz(B_CNT * DIM_C, 256), 256, 0, stream>>>(hl0, 0.0f, B_CNT * DIM_C);
  fill_f32<<<gsz(B_CNT * DIM_C, 256), 256, 0, stream>>>(cl, 0.0f, B_CNT * DIM_C);
  fill_f32<<<gsz(B_CNT * 2 * DIM_C, 256), 256, 0, stream>>>(qstar, 0.0f, B_CNT * 2 * DIM_C);
  float* hl_in = hl0;
  float* hl_out = hl1;
  for (int step = 0; step < 3; ++step) {
    lstm_kernel<<<gsz(B_CNT * DIM_C, 256), 256, 0, stream>>>(qstar, hl_in, hl_out, cl,
                                                             lwih, lwhh, lbih, lbhh);
    fill_f32<<<gsz(B_CNT, 256), 256, 0, stream>>>(emax, -INFINITY, B_CNT);
    fill_f32<<<gsz(B_CNT, 256), 256, 0, stream>>>(esum, 0.0f, B_CNT);
    fill_f32<<<gsz(B_CNT * DIM_C, 256), 256, 0, stream>>>(rbuf, 0.0f, B_CNT * DIM_C);
    attn_e_kernel<<<gsz(N_CNT, 256), 256, 0, stream>>>(h, hl_out, batch, ebuf, emax);
    attn_exp_kernel<<<gsz(N_CNT, 256), 256, 0, stream>>>(ebuf, emax, batch, exbuf, esum);
    attn_r_kernel<<<gsz((long long)N_CNT * DIM_C, 256), 256, 0, stream>>>(h, exbuf, esum, batch, rbuf);
    qstar_kernel<<<gsz(B_CNT * 2 * DIM_C, 256), 256, 0, stream>>>(hl_out, rbuf, qstar);
    float* tmp = hl_in; hl_in = hl_out; hl_out = tmp;
  }

  // ---- final MLP ----
  final_kernel<<<gsz(B_CNT, 256), 256, 0, stream>>>(qstar, lin1_w, lin1_b, lin2_w, lin2_b, out);
}
